// FP8PureLinear_58806692216899
// MI455X (gfx1250) — 8-bit matrix operand order repaired and checked by reading, compiled, not run on hardware
//
#include <hip/hip_runtime.h>
#include <hip/hip_bf16.h>

typedef __attribute__((ext_vector_type(16))) int   v16i;
typedef __attribute__((ext_vector_type(8)))  float v8f;

#define E4M3_MAX 448.0f

#if __has_builtin(__builtin_amdgcn_global_load_async_to_lds_b128)
#define HAVE_ASYNC_LDS 1
typedef int v4i_vs __attribute__((vector_size(16)));
typedef __attribute__((address_space(1))) v4i_vs as1_v4i;
typedef __attribute__((address_space(3))) v4i_vs as3_v4i;
static __device__ __forceinline__ void async_copy_b128(const void* gptr, void* lptr) {
    __builtin_amdgcn_global_load_async_to_lds_b128((as1_v4i*)gptr, (as3_v4i*)lptr,
                                                   0, 0);
}
static __device__ __forceinline__ void wait_async_copies() {
#if __has_builtin(__builtin_amdgcn_s_wait_asynccnt)
    __builtin_amdgcn_s_wait_asynccnt(0);
#else
    asm volatile("s_wait_asynccnt 0x0" ::: "memory");
#endif
}
#else
#define HAVE_ASYNC_LDS 0
static __device__ __forceinline__ void async_copy_b128(const void* gptr, void* lptr) {
    *(uint4*)lptr = *(const uint4*)gptr;
}
static __device__ __forceinline__ void wait_async_copies() {}
#endif

static __device__ __forceinline__ unsigned char f32_to_e4m3_sw(float x) {
    unsigned u = __float_as_uint(x);
    unsigned s = (u >> 24) & 0x80u;
    float a = fminf(fabsf(x), E4M3_MAX);
    if (!(a >= 0x1p-10f)) return (unsigned char)s;
    int e;
    float m = frexpf(a, &e);
    int eb = (e - 1) + 7;
    unsigned bits;
    if (eb <= 0) {
        bits = (unsigned)(int)rintf(a * 512.0f);
    } else {
        int man = (int)rintf(m * 16.0f) - 8;
        if (man == 8) { man = 0; ++eb; }
        bits = ((unsigned)eb << 3) | (unsigned)(man & 7);
    }
    if (bits > 0x7Eu) bits = 0x7Eu;
    return (unsigned char)(s | bits);
}

static __device__ __forceinline__ unsigned pack4_e4m3(float f0, float f1,
                                                      float f2, float f3) {
#if __has_builtin(__builtin_amdgcn_cvt_pk_fp8_f32)
    int r = __builtin_amdgcn_cvt_pk_fp8_f32(f0, f1, 0, false);
    r = __builtin_amdgcn_cvt_pk_fp8_f32(f2, f3, r, true);
    return (unsigned)r;
#else
    return (unsigned)f32_to_e4m3_sw(f0) |
           ((unsigned)f32_to_e4m3_sw(f1) << 8) |
           ((unsigned)f32_to_e4m3_sw(f2) << 16) |
           ((unsigned)f32_to_e4m3_sw(f3) << 24);
#endif
}

static __device__ __forceinline__ float bfbits_to_f32(unsigned h) {
    return __uint_as_float(h << 16);
}

__global__ void init_amax_kernel(unsigned* amax_bits) {
    if (threadIdx.x == 0 && blockIdx.x == 0) *amax_bits = 0u;
}

__global__ void amax_kernel(const unsigned short* __restrict__ x, size_t n,
                            unsigned* __restrict__ amax_bits) {
    size_t stride = (size_t)gridDim.x * blockDim.x * 8;
    size_t i = ((size_t)blockIdx.x * blockDim.x + threadIdx.x) * 8;
    unsigned m = 0u;
    for (; i < n; i += stride) {
        uint4 d = *(const uint4*)(x + i);
        unsigned w0 = d.x, w1 = d.y, w2 = d.z, w3 = d.w;
        m = max(m, (w0 & 0x7FFFu)); m = max(m, ((w0 >> 16) & 0x7FFFu));
        m = max(m, (w1 & 0x7FFFu)); m = max(m, ((w1 >> 16) & 0x7FFFu));
        m = max(m, (w2 & 0x7FFFu)); m = max(m, ((w2 >> 16) & 0x7FFFu));
        m = max(m, (w3 & 0x7FFFu)); m = max(m, ((w3 >> 16) & 0x7FFFu));
    }
    #pragma unroll
    for (int off = 16; off > 0; off >>= 1)
        m = max(m, (unsigned)__shfl_xor((int)m, off, 32));
    if ((threadIdx.x & 31) == 0)
        atomicMax(amax_bits, m << 16);
}

__global__ void quantize_x_kernel(const unsigned short* __restrict__ x,
                                  unsigned char* __restrict__ xq, size_t n,
                                  const unsigned* __restrict__ amax_bits) {
    float amax = __uint_as_float(*amax_bits);
    float inv_s = 1.0f / fmaxf(amax / E4M3_MAX, 1e-12f);
    size_t stride = (size_t)gridDim.x * blockDim.x * 8;
    size_t i = ((size_t)blockIdx.x * blockDim.x + threadIdx.x) * 8;
    for (; i < n; i += stride) {
        uint4 d = *(const uint4*)(x + i);
        float f0 = bfbits_to_f32(d.x & 0xFFFFu) * inv_s;
        float f1 = bfbits_to_f32(d.x >> 16) * inv_s;
        float f2 = bfbits_to_f32(d.y & 0xFFFFu) * inv_s;
        float f3 = bfbits_to_f32(d.y >> 16) * inv_s;
        float f4 = bfbits_to_f32(d.z & 0xFFFFu) * inv_s;
        float f5 = bfbits_to_f32(d.z >> 16) * inv_s;
        float f6 = bfbits_to_f32(d.w & 0xFFFFu) * inv_s;
        float f7 = bfbits_to_f32(d.w >> 16) * inv_s;
        uint2 o;
        o.x = pack4_e4m3(f0, f1, f2, f3);
        o.y = pack4_e4m3(f4, f5, f6, f7);
        *(uint2*)(xq + i) = o;
    }
}

__global__ void quantize_w_kernel(const float* __restrict__ w,
                                  unsigned char* __restrict__ wq, size_t n) {
    size_t stride = (size_t)gridDim.x * blockDim.x * 8;
    size_t i = ((size_t)blockIdx.x * blockDim.x + threadIdx.x) * 8;
    for (; i < n; i += stride) {
        float4 a = *(const float4*)(w + i);
        float4 b = *(const float4*)(w + i + 4);
        uint2 o;
        o.x = pack4_e4m3(a.x, a.y, a.z, a.w);
        o.y = pack4_e4m3(b.x, b.y, b.z, b.w);
        *(uint2*)(wq + i) = o;
    }
}

__global__ __launch_bounds__(256, 1)
void fp8_gemm_kernel(const unsigned char* __restrict__ Aq,
                     const unsigned char* __restrict__ Wq,
                     const unsigned* __restrict__ amax_bits,
                     const float* __restrict__ wscale,
                     const __hip_bfloat16* __restrict__ bias,
                     __hip_bfloat16* __restrict__ out,
                     int M, int N, int K) {
    __shared__ __align__(16) unsigned char ldsA[2][128 * 128];
    __shared__ __align__(16) unsigned char ldsB[2][128 * 128];

    const int lane  = threadIdx.x & 31;
    const int wave  = threadIdx.x >> 5;
    const int wm    = wave >> 1;
    const int wn    = wave & 1;
    const int tileMBase = blockIdx.y * 128;
    const int tileNBase = blockIdx.x * 128;
    const int l16   = lane & 15;
    const int half  = lane >> 4;

    const int t    = threadIdx.x;
    const int lrow = t >> 1;
    const int lcol = (t & 1) * 64;
    const unsigned char* gA = Aq + (size_t)(tileMBase + lrow) * K + lcol;
    const unsigned char* gB = Wq + (size_t)(tileNBase + lrow) * K + lcol;
    const int lofs = lrow * 128 + lcol;

    const int aofs0 = (wm * 32 + l16) * 128 + half * 16;
    const int aofs1 = aofs0 + 16 * 128;
    int bofs[4];
    #pragma unroll
    for (int jf = 0; jf < 4; ++jf)
        bofs[jf] = (wn * 64 + jf * 16 + l16) * 128 + half * 16;

    v8f acc[2][4];
    #pragma unroll
    for (int i = 0; i < 2; ++i)
        #pragma unroll
        for (int jf = 0; jf < 4; ++jf)
            #pragma unroll
            for (int r = 0; r < 8; ++r)
                acc[i][jf][r] = 0.0f;

    const int nk = K >> 7;

    #pragma unroll
    for (int c = 0; c < 4; ++c) {
        async_copy_b128(gA + 16 * c, &ldsA[0][lofs + 16 * c]);
        async_copy_b128(gB + 16 * c, &ldsB[0][lofs + 16 * c]);
    }

    for (int ki = 0; ki < nk; ++ki) {
        wait_async_copies();
        __syncthreads();
        const int cur = ki & 1;

        if (ki + 1 < nk) {
            const int nxt = cur ^ 1;
            const int kb  = (ki + 1) << 7;
            #pragma unroll
            for (int c = 0; c < 4; ++c) {
                async_copy_b128(gA + kb + 16 * c, &ldsA[nxt][lofs + 16 * c]);
                async_copy_b128(gB + kb + 16 * c, &ldsB[nxt][lofs + 16 * c]);
            }
        }

        const unsigned char* lA = ldsA[cur];
        const unsigned char* lB = ldsB[cur];
        v16i a0, a1;
        #pragma unroll
        for (int j = 0; j < 8; ++j) {
            uint2 d0 = *(const uint2*)(lA + aofs0 + 32 * (j >> 1) + 8 * (j & 1));
            uint2 d1 = *(const uint2*)(lA + aofs1 + 32 * (j >> 1) + 8 * (j & 1));
            a0[2 * j] = (int)d0.x; a0[2 * j + 1] = (int)d0.y;
            a1[2 * j] = (int)d1.x; a1[2 * j + 1] = (int)d1.y;
        }
        v16i b[4];
        #pragma unroll
        for (int jf = 0; jf < 4; ++jf) {
            #pragma unroll
            for (int g = 0; g < 4; ++g) {
                uint4 d = *(const uint4*)(lB + bofs[jf] + 32 * g);
                b[jf][4 * g + 0] = (int)d.x;
                b[jf][4 * g + 1] = (int)d.y;
                b[jf][4 * g + 2] = (int)d.z;
                b[jf][4 * g + 3] = (int)d.w;
            }
        }
        #pragma unroll
        for (int jf = 0; jf < 4; ++jf) {
            acc[0][jf] = __builtin_amdgcn_wmma_f32_16x16x128_fp8_fp8(
                a0, b[jf], (short)0, acc[0][jf], false, false);
            acc[1][jf] = __builtin_amdgcn_wmma_f32_16x16x128_fp8_fp8(
                a1, b[jf], (short)0, acc[1][jf], false, false);
        }
    }

    float amax  = __uint_as_float(*amax_bits);
    float s     = fmaxf(amax / E4M3_MAX, 1e-12f);
    float total = s * wscale[0];

    #pragma unroll
    for (int jf = 0; jf < 4; ++jf) {
        int col = tileNBase + wn * 64 + jf * 16 + l16;
        float bz = __bfloat162float(bias[col]);
        #pragma unroll
        for (int i = 0; i < 2; ++i) {
            int rbase = tileMBase + wm * 32 + i * 16 + half * 8;
            #pragma unroll
            for (int r = 0; r < 8; ++r) {
                float v = acc[i][jf][r] * total;
                float vb = __bfloat162float(__float2bfloat16(v));
                out[(size_t)(rbase + r) * N + col] = __float2bfloat16(vb + bz);
            }
        }
    }
}

extern "C" void kernel_launch(void* const* d_in, const int* in_sizes, int n_in,
                              void* d_out, int out_size, void* d_ws, size_t ws_size,
                              hipStream_t stream) {
    const unsigned short*  x      = (const unsigned short*)d_in[0];
    const float*           wq_f32 = (const float*)d_in[1];
    const float*           wscale = (const float*)d_in[2];
    const __hip_bfloat16*  bias   = (const __hip_bfloat16*)d_in[3];
    __hip_bfloat16*        out    = (__hip_bfloat16*)d_out;

    const int    N  = in_sizes[3];
    const int    K  = in_sizes[1] / N;
    const size_t MK = (size_t)in_sizes[0];
    const int    M  = (int)(MK / (size_t)K);
    const size_t NK = (size_t)N * (size_t)K;

    unsigned*      amax_bits = (unsigned*)d_ws;
    unsigned char* xq        = (unsigned char*)d_ws + 256;
    unsigned char* wq8       = xq + MK;

    init_amax_kernel<<<1, 1, 0, stream>>>(amax_bits);
    amax_kernel<<<2048, 256, 0, stream>>>(x, MK, amax_bits);
    quantize_x_kernel<<<4096, 256, 0, stream>>>(x, xq, MK, amax_bits);
    quantize_w_kernel<<<2048, 256, 0, stream>>>(wq_f32, wq8, NK);

    dim3 grid(N / 128, M / 128);
    fp8_gemm_kernel<<<grid, 256, 0, stream>>>(xq, wq8, amax_bits, wscale, bias,
                                              out, M, N, K);
}
